// CausalSelfAttention_7009386627325
// MI455X (gfx1250) — matrix operand order repaired and checked by reading, not compiled, not run on hardware
//
#include <hip/hip_runtime.h>
#include <cstdint>
#include <cstddef>

// ---------------------------------------------------------------------------
// CDNA5 (gfx1250, wave32) causal self-attention forward.
// Matmuls: v_wmma_f32_16x16x32_bf16 (f32 accumulate).
// Staging: global_load_async_to_lds_b128 + double-buffered LDS prefetch.
// ---------------------------------------------------------------------------

typedef __attribute__((ext_vector_type(16))) __bf16 bf16x16;
typedef __attribute__((ext_vector_type(8)))  __bf16 bf16x8;
typedef __attribute__((ext_vector_type(8)))  float  f32x8;

#define N_HEADC   16
#define HEAD_DIMC 64
#define N_EMBDC   1024
#define SEQC      2048
#define BATCHC    4
#define M_TOTC    (BATCHC * SEQC)       // 8192
#define SCALEC    0.125f                // 1/sqrt(64)

// ---------------- fragment helpers ----------------

__device__ __forceinline__ bf16x8 ld8(const __bf16* p) {
  return *reinterpret_cast<const bf16x8*>(p);
}
__device__ __forceinline__ bf16x16 cat16(bf16x8 lo, bf16x8 hi) {
  return __builtin_shufflevector(lo, hi, 0,1,2,3,4,5,6,7,8,9,10,11,12,13,14,15);
}

// A fragment (16x32 bf16, row-major source, leading dim ld):
//   lane<16 : row=lane,    K = {0..7, 16..23}
//   lane>=16: row=lane-16, K = {8..15, 24..31}
__device__ __forceinline__ bf16x16 load_a_frag(const __bf16* tile, int ld, int lane) {
  const __bf16* r = tile + (size_t)(lane & 15) * ld + ((lane >> 4) << 3);
  return cat16(ld8(r), ld8(r + 16));
}

// B fragment (32x16 bf16) from transposed operand Bt = [N][K] row-major:
//   lane<16 : col=lane,    K = {0..7, 16..23}
//   lane>=16: col=lane-16, K = {8..15, 24..31}
__device__ __forceinline__ bf16x16 load_b_frag(const __bf16* tileT, int ld, int lane) {
  const __bf16* r = tileT + (size_t)(lane & 15) * ld + ((lane >> 4) << 3);
  return cat16(ld8(r), ld8(r + 16));
}

__device__ __forceinline__ f32x8 wmma_bf16(bf16x16 a, bf16x16 b, f32x8 c) {
  return __builtin_amdgcn_wmma_f32_16x16x32_bf16(false, a, false, b,
                                                 short(0), c, false, false);
}

// ---------------- async copy helpers (CDNA5 TDM-lite path) ----------------
// 16B HBM -> LDS with no VGPR staging, tracked by ASYNCcnt.
// LDS destination: flat shared pointer truncated to 32 bits == LDS offset
// (see ISA section 10.2).

__device__ __forceinline__ void async_cp16(const __bf16* lds_dst, const __bf16* gsrc) {
  unsigned      ldsoff = (unsigned)(size_t)lds_dst;
  unsigned long long ga = (unsigned long long)(size_t)gsrc;
  asm volatile("global_load_async_to_lds_b128 %0, %1, off"
               :: "v"(ldsoff), "v"(ga) : "memory");
}
__device__ __forceinline__ void async_wait0() {
  asm volatile("s_wait_asynccnt 0" ::: "memory");
}

// ---------------- conversion kernels ----------------

__global__ __launch_bounds__(256) void cvt_f32_bf16(const float* __restrict__ src,
                                                    __bf16* __restrict__ dst, int n) {
  int i = blockIdx.x * 256 + threadIdx.x;
  if (i < n) dst[i] = (__bf16)src[i];
}

// W [K][N] f32  ->  Wt [N][K] bf16  (coalesced writes, strided reads)
__global__ __launch_bounds__(256) void cvt_transpose(const float* __restrict__ W,
                                                     __bf16* __restrict__ Wt,
                                                     int K, int N) {
  int i = blockIdx.x * 256 + threadIdx.x;   // i indexes Wt flat = n*K + k
  if (i < K * N) {
    int n = i / K, k = i - n * K;
    Wt[i] = (__bf16)W[(size_t)k * N + n];
  }
}

// ---------------- QKV GEMM ----------------
// A = x_bf16 [8192][1024] row-major, Bt = Wattn^T [3072][1024] row-major.
// Output scattered into q[b,h,t,d], k[b,h,t,d], vt[b,h,d,t] (bf16, +bias).

#define GLDA 40   // padded LDS leading dim (32 + 8) to break bank conflicts

// issue async loads of one 128x32 tile (padded) for a 256-thread block
__device__ __forceinline__ void stage_tile_128x32(__bf16* sdst, const __bf16* gsrc,
                                                  size_t gld, int tid) {
#pragma unroll
  for (int it = 0; it < 2; ++it) {
    int c = tid + it * 256;              // 512 chunks of 8 elems
    int row = c >> 2, off = (c & 3) * 8;
    async_cp16(&sdst[row * GLDA + off], &gsrc[(size_t)row * gld + off]);
  }
}

__global__ __launch_bounds__(256) void gemm_qkv_kernel(const __bf16* __restrict__ A,
                                                       const __bf16* __restrict__ Bt,
                                                       const float*  __restrict__ bias,
                                                       __bf16* __restrict__ q,
                                                       __bf16* __restrict__ k,
                                                       __bf16* __restrict__ vt) {
  __shared__ __bf16 sA[2][128 * GLDA];
  __shared__ __bf16 sB[2][128 * GLDA];

  const int K = N_EMBDC;                    // 1024
  const int bm = blockIdx.y * 128;
  const int bn = blockIdx.x * 128;
  const int tid = threadIdx.x, lane = tid & 31, wid = tid >> 5;
  const int wm = (wid & 3) * 32;            // 4 waves along M
  const int wn = (wid >> 2) * 64;           // 2 waves along N
  const int half = lane >> 4;

  f32x8 acc[2][4] = {};

  // prologue: prefetch tile 0
  stage_tile_128x32(sA[0], &A[(size_t)bm * K], K, tid);
  stage_tile_128x32(sB[0], &Bt[(size_t)bn * K], K, tid);
  async_wait0();
  __syncthreads();

  const int NK = K / 32;
  for (int ki = 0; ki < NK; ++ki) {
    int cur = ki & 1;
    if (ki + 1 < NK) {                      // prefetch next tile into other buffer
      int k1 = (ki + 1) * 32;
      stage_tile_128x32(sA[cur ^ 1], &A[(size_t)bm * K + k1], K, tid);
      stage_tile_128x32(sB[cur ^ 1], &Bt[(size_t)bn * K + k1], K, tid);
    }

    bf16x16 af[2], bfr[4];
#pragma unroll
    for (int i = 0; i < 2; i++) af[i] = load_a_frag(&sA[cur][(wm + i * 16) * GLDA], GLDA, lane);
#pragma unroll
    for (int j = 0; j < 4; j++) bfr[j] = load_b_frag(&sB[cur][(wn + j * 16) * GLDA], GLDA, lane);
#pragma unroll
    for (int i = 0; i < 2; i++)
#pragma unroll
      for (int j = 0; j < 4; j++)
        acc[i][j] = wmma_bf16(af[i], bfr[j], acc[i][j]);

    async_wait0();                          // next tile landed
    __syncthreads();                        // everyone done reading cur
  }

  // scatter epilogue: n<1024 -> q, <2048 -> k, else v transposed
#pragma unroll
  for (int i = 0; i < 2; i++) {
#pragma unroll
    for (int j = 0; j < 4; j++) {
      int n = bn + wn + j * 16 + (lane & 15);
      float bv = bias[n];
      int which = n >> 10;            // 0:q 1:k 2:v
      int nn = n & 1023;
      int h = nn >> 6, d = nn & 63;
#pragma unroll
      for (int r = 0; r < 8; r++) {
        int m = bm + wm + i * 16 + r + 8 * half;   // m = b*T + t
        int b = m >> 11, t = m & (SEQC - 1);
        size_t bh = (size_t)(b * N_HEADC + h);
        __bf16 v = (__bf16)(acc[i][j][r] + bv);
        if (which == 0)       q [(bh * SEQC + t) * HEAD_DIMC + d] = v;
        else if (which == 1)  k [(bh * SEQC + t) * HEAD_DIMC + d] = v;
        else                  vt[(bh * HEAD_DIMC + d) * SEQC + t] = v;
      }
    }
  }
}

// ---------------- flash attention ----------------
// grid: (T/64, B*H), 128 threads = 4 waves; wave w owns queries q0..q0+15.
// K block [64 keys][64 d] and Vt block [64 d][64 keys] staged in LDS (async,
// double-buffered across key blocks).

#define ALDA 72   // padded LDS leading dim (64 + 8)

__device__ __forceinline__ void stage_kv_block(__bf16* sk, __bf16* sv,
                                               const __bf16* Kg, const __bf16* Vg,
                                               int tid) {
#pragma unroll
  for (int it = 0; it < 4; ++it) {         // 512 chunks over 128 threads
    int c = tid + it * 128;
    int row = c >> 3, off = (c & 7) * 8;
    async_cp16(&sk[row * ALDA + off], &Kg[(size_t)row * HEAD_DIMC + off]);
  }
#pragma unroll
  for (int it = 0; it < 4; ++it) {
    int c = tid + it * 128;
    int row = c >> 3, off = (c & 7) * 8;
    async_cp16(&sv[row * ALDA + off], &Vg[(size_t)row * SEQC + off]);
  }
}

__global__ __launch_bounds__(128) void attn_kernel(const __bf16* __restrict__ qg,
                                                   const __bf16* __restrict__ kg,
                                                   const __bf16* __restrict__ vtg,
                                                   __bf16* __restrict__ y) {
  __shared__ __bf16 sK[2][64 * ALDA];
  __shared__ __bf16 sV[2][64 * ALDA];
  __shared__ __bf16 sP[4][16 * ALDA];

  const int bh = blockIdx.y;                 // 0..63
  const int qb = blockIdx.x;                 // 0..31
  const int tid = threadIdx.x, lane = tid & 31, w = tid >> 5;
  const int half = lane >> 4;
  const int q0 = qb * 64 + w * 16;

  const __bf16* Kbase = kg  + (size_t)bh * SEQC * HEAD_DIMC;
  const __bf16* Vbase = vtg + (size_t)bh * HEAD_DIMC * SEQC;

  // preload Q fragments for the whole d=64 (two K-chunks), straight from HBM
  bf16x16 aq[2];
  {
    const __bf16* r0 = qg + ((size_t)bh * SEQC + q0 + (lane & 15)) * HEAD_DIMC + half * 8;
    aq[0] = cat16(ld8(r0),      ld8(r0 + 16));
    aq[1] = cat16(ld8(r0 + 32), ld8(r0 + 48));
  }

  f32x8 o[4] = {};
  float mi[8], li[8];
#pragma unroll
  for (int r = 0; r < 8; r++) { mi[r] = -1e30f; li[r] = 0.0f; }

  // prologue: prefetch key block 0
  stage_kv_block(sK[0], sV[0], Kbase, Vbase, tid);
  async_wait0();
  __syncthreads();

  for (int jb = 0; jb <= qb; ++jb) {
    int cur = jb & 1;
    if (jb + 1 <= qb)
      stage_kv_block(sK[cur ^ 1], sV[cur ^ 1],
                     Kbase + (size_t)(jb + 1) * 64 * HEAD_DIMC,
                     Vbase + (size_t)(jb + 1) * 64, tid);

    // S = Q K^T  (4 key sub-tiles of 16)
    f32x8 st[4];
#pragma unroll
    for (int nt = 0; nt < 4; nt++) {
      f32x8 s = {};
#pragma unroll
      for (int c = 0; c < 2; c++) {
        bf16x16 b = load_b_frag(&sK[cur][(nt * 16) * ALDA + c * 32], ALDA, lane);
        s = wmma_bf16(aq[c], b, s);
      }
      st[nt] = s;
    }

    const bool diag = (jb == qb);
    float rm[8];
#pragma unroll
    for (int r = 0; r < 8; r++) rm[r] = -1e30f;
#pragma unroll
    for (int nt = 0; nt < 4; nt++) {
      int key = jb * 64 + nt * 16 + (lane & 15);
#pragma unroll
      for (int r = 0; r < 8; r++) {
        float s = st[nt][r] * SCALEC;
        if (diag && key > q0 + r + 8 * half) s = -1e30f;
        st[nt][r] = s;
        rm[r] = fmaxf(rm[r], s);
      }
    }
    // row-max across the 16 lanes of each half-wave
#pragma unroll
    for (int msk = 1; msk < 16; msk <<= 1)
#pragma unroll
      for (int r = 0; r < 8; r++) rm[r] = fmaxf(rm[r], __shfl_xor(rm[r], msk, 32));

    float alpha[8], rs[8];
#pragma unroll
    for (int r = 0; r < 8; r++) {
      float nm = fmaxf(mi[r], rm[r]);
      alpha[r] = __expf(mi[r] - nm);
      mi[r] = nm;
      rs[r] = 0.0f;
    }

    // P = exp(S - m), row-sum, park P in per-wave LDS scratch (C-layout -> A-layout)
#pragma unroll
    for (int nt = 0; nt < 4; nt++) {
      int col = nt * 16 + (lane & 15);
#pragma unroll
      for (int r = 0; r < 8; r++) {
        float p = __expf(st[nt][r] - mi[r]);
        rs[r] += p;
        sP[w][(r + 8 * half) * ALDA + col] = (__bf16)p;
      }
    }
#pragma unroll
    for (int msk = 1; msk < 16; msk <<= 1)
#pragma unroll
      for (int r = 0; r < 8; r++) rs[r] += __shfl_xor(rs[r], msk, 32);
#pragma unroll
    for (int r = 0; r < 8; r++) li[r] = li[r] * alpha[r] + rs[r];
#pragma unroll
    for (int nd = 0; nd < 4; nd++)
#pragma unroll
      for (int r = 0; r < 8; r++) o[nd][r] *= alpha[r];

    // reload P as A fragments (same-wave LDS RAW: DS ops are in-order per wave)
    bf16x16 ap[2];
    {
      const __bf16* r0 = &sP[w][(lane & 15) * ALDA + half * 8];
      ap[0] = cat16(ld8(r0),      ld8(r0 + 16));
      ap[1] = cat16(ld8(r0 + 32), ld8(r0 + 48));
    }
    // O += P * V   (V supplied transposed: Bt rows are output dims d)
#pragma unroll
    for (int nd = 0; nd < 4; nd++)
#pragma unroll
      for (int c = 0; c < 2; c++) {
        bf16x16 b = load_b_frag(&sV[cur][(nd * 16) * ALDA + c * 32], ALDA, lane);
        o[nd] = wmma_bf16(ap[c], b, o[nd]);
      }

    async_wait0();                           // next K/V block landed
    __syncthreads();                         // all waves done reading cur
  }

  // epilogue: O / l  ->  y[b,t, h*64+d] (bf16, row-major A for the proj GEMM)
  const int b = bh >> 4, h = bh & 15;
#pragma unroll
  for (int nd = 0; nd < 4; nd++) {
    int d = nd * 16 + (lane & 15);
#pragma unroll
    for (int r = 0; r < 8; r++) {
      int t = q0 + r + 8 * half;
      float v = o[nd][r] / li[r];
      y[((size_t)b * SEQC + t) * N_EMBDC + h * HEAD_DIMC + d] = (__bf16)v;
    }
  }
}

// ---------------- output projection GEMM ----------------
// A = y_bf16 [8192][1024], Bt = Wproj^T [1024][1024], out f32 [8192][1024] + bias.

__global__ __launch_bounds__(256) void gemm_proj_kernel(const __bf16* __restrict__ A,
                                                        const __bf16* __restrict__ Bt,
                                                        const float*  __restrict__ bias,
                                                        float* __restrict__ out) {
  __shared__ __bf16 sA[2][128 * GLDA];
  __shared__ __bf16 sB[2][128 * GLDA];

  const int K = N_EMBDC;
  const int bm = blockIdx.y * 128;
  const int bn = blockIdx.x * 128;
  const int tid = threadIdx.x, lane = tid & 31, wid = tid >> 5;
  const int wm = (wid & 3) * 32;
  const int wn = (wid >> 2) * 64;
  const int half = lane >> 4;

  f32x8 acc[2][4] = {};

  stage_tile_128x32(sA[0], &A[(size_t)bm * K], K, tid);
  stage_tile_128x32(sB[0], &Bt[(size_t)bn * K], K, tid);
  async_wait0();
  __syncthreads();

  const int NK = K / 32;
  for (int ki = 0; ki < NK; ++ki) {
    int cur = ki & 1;
    if (ki + 1 < NK) {
      int k1 = (ki + 1) * 32;
      stage_tile_128x32(sA[cur ^ 1], &A[(size_t)bm * K + k1], K, tid);
      stage_tile_128x32(sB[cur ^ 1], &Bt[(size_t)bn * K + k1], K, tid);
    }

    bf16x16 af[2], bfr[4];
#pragma unroll
    for (int i = 0; i < 2; i++) af[i] = load_a_frag(&sA[cur][(wm + i * 16) * GLDA], GLDA, lane);
#pragma unroll
    for (int j = 0; j < 4; j++) bfr[j] = load_b_frag(&sB[cur][(wn + j * 16) * GLDA], GLDA, lane);
#pragma unroll
    for (int i = 0; i < 2; i++)
#pragma unroll
      for (int j = 0; j < 4; j++)
        acc[i][j] = wmma_bf16(af[i], bfr[j], acc[i][j]);

    async_wait0();
    __syncthreads();
  }

#pragma unroll
  for (int i = 0; i < 2; i++)
#pragma unroll
    for (int j = 0; j < 4; j++) {
      int n = bn + wn + j * 16 + (lane & 15);
      float bv = bias[n];
#pragma unroll
      for (int r = 0; r < 8; r++) {
        int m = bm + wm + i * 16 + r + 8 * half;
        out[(size_t)m * N_EMBDC + n] = acc[i][j][r] + bv;
      }
    }
}

// ---------------- host launcher ----------------

extern "C" void kernel_launch(void* const* d_in, const int* in_sizes, int n_in,
                              void* d_out, int out_size, void* d_ws, size_t ws_size,
                              hipStream_t stream) {
  const float* x      = (const float*)d_in[0];   // [4,2048,1024]
  const float* W_attn = (const float*)d_in[1];   // [1024,3072]
  const float* b_attn = (const float*)d_in[2];   // [3072]
  const float* W_proj = (const float*)d_in[3];   // [1024,1024]
  const float* b_proj = (const float*)d_in[4];   // [1024]
  float* out = (float*)d_out;

  // workspace carve-up (bf16 elements), ~88 MiB total
  __bf16* xb  = (__bf16*)d_ws;
  __bf16* wa  = xb  + (size_t)M_TOTC * N_EMBDC;            // 8192*1024
  __bf16* wp  = wa  + (size_t)3 * N_EMBDC * N_EMBDC;       // 3072*1024
  __bf16* qb  = wp  + (size_t)N_EMBDC * N_EMBDC;           // 1024*1024
  __bf16* kb  = qb  + (size_t)M_TOTC * N_EMBDC;
  __bf16* vtb = kb  + (size_t)M_TOTC * N_EMBDC;
  __bf16* yb  = vtb + (size_t)M_TOTC * N_EMBDC;

  int nx = M_TOTC * N_EMBDC;
  cvt_f32_bf16<<<(nx + 255) / 256, 256, 0, stream>>>(x, xb, nx);
  int nwa = N_EMBDC * 3 * N_EMBDC;
  cvt_transpose<<<(nwa + 255) / 256, 256, 0, stream>>>(W_attn, wa, N_EMBDC, 3 * N_EMBDC);
  int nwp = N_EMBDC * N_EMBDC;
  cvt_transpose<<<(nwp + 255) / 256, 256, 0, stream>>>(W_proj, wp, N_EMBDC, N_EMBDC);

  gemm_qkv_kernel<<<dim3(3 * N_EMBDC / 128, M_TOTC / 128), 256, 0, stream>>>(
      xb, wa, b_attn, qb, kb, vtb);

  attn_kernel<<<dim3(SEQC / 64, BATCHC * N_HEADC), 128, 0, stream>>>(qb, kb, vtb, yb);

  gemm_proj_kernel<<<dim3(N_EMBDC / 128, M_TOTC / 128), 256, 0, stream>>>(
      yb, wp, b_proj, out);
}
